// MultilayerSNN_33560874451070
// MI455X (gfx1250) — 16-bit and 8-bit matrix operand order repaired and checked by reading, compiled, not run on hardware
//
#include <hip/hip_runtime.h>


typedef __attribute__((ext_vector_type(16))) _Float16 v16h;
typedef __attribute__((ext_vector_type(8)))  _Float16 v8h;
typedef __attribute__((ext_vector_type(8)))  float    v8f;
typedef __attribute__((ext_vector_type(8)))  int      v8i;
typedef __attribute__((ext_vector_type(4)))  int      v4i;
typedef __attribute__((ext_vector_type(2)))  int      v2i;

#define T_STEPS 50
#define D_IN    256
#define H1      1024
#define H2      1024
#define D_OUT   512
#define MTILE   16
#define THREADS 256

#define LOAD_A8(dst, rowptr, koff) do {                                        \
    v2i _c0 = *(const v2i*)&(rowptr)[(koff)];                                  \
    v2i _c1 = *(const v2i*)&(rowptr)[(koff) + 8];                              \
    v2i _c2 = *(const v2i*)&(rowptr)[(koff) + 32];                             \
    v2i _c3 = *(const v2i*)&(rowptr)[(koff) + 40];                             \
    dst[0] = _c0[0]; dst[1] = _c0[1]; dst[2] = _c1[0]; dst[3] = _c1[1];        \
    dst[4] = _c2[0]; dst[5] = _c2[1]; dst[6] = _c3[0]; dst[7] = _c3[1];        \
} while (0)

#define LOAD_B8(dst, wr, ks) do {                                              \
    v4i _lo = *(const v4i*)((wr) + (ks));                                      \
    v4i _hi = *(const v4i*)((wr) + (ks) + 32);                                 \
    dst[0] = _lo[0]; dst[1] = _lo[1]; dst[2] = _lo[2]; dst[3] = _lo[3];        \
    dst[4] = _hi[0]; dst[5] = _hi[1]; dst[6] = _hi[2]; dst[7] = _hi[3];        \
} while (0)

#define WMMA_IU8(a, b, c)                                                      \
    __builtin_amdgcn_wmma_i32_16x16x64_iu8(false, (a), false, (b), (c), false, false)
#define WMMA_F16(a, b, c)                                                      \
    __builtin_amdgcn_wmma_f32_16x16x32_f16(false, (a), false, (b), (short)0, (c), false, false)


__global__ void cvt_f32_to_f16_kernel(const float* __restrict__ in,
                                      _Float16* __restrict__ out, int n) {
    int i = blockIdx.x * blockDim.x + threadIdx.x;
    if (i < n) out[i] = (_Float16)in[i];
}

__global__ void cvt_f32_to_u8_kernel(const float* __restrict__ in,
                                     unsigned char* __restrict__ out, int n) {
    int i = blockIdx.x * blockDim.x + threadIdx.x;
    if (i < n) out[i] = (unsigned char)(in[i] != 0.0f ? 1 : 0);
}


__global__ __launch_bounds__(THREADS)
void snn_fused_kernel(const float* __restrict__ xin,
                      const _Float16* __restrict__ W0h,
                      const unsigned char* __restrict__ W1b,
                      const unsigned char* __restrict__ W2b,
                      float* __restrict__ out)
{
    __shared__ __attribute__((aligned(16))) _Float16      xs[MTILE][D_IN];
    __shared__ __attribute__((aligned(16))) unsigned char s0[MTILE][H1];
    __shared__ __attribute__((aligned(16))) unsigned char s1[MTILE][H2];

    const int tid  = threadIdx.x;
    const int wave = tid >> 5;
    const int lane = tid & 31;
    const int lm   = lane & 15;
    const int lhi  = lane >> 4;
    const int b0   = blockIdx.x * MTILE;

    v8f mem0[8];
    v8f mem1[8];
    v8f mem2[4];
    #pragma unroll
    for (int i = 0; i < 8; ++i) {
        #pragma unroll
        for (int j = 0; j < 8; ++j) { mem0[i][j] = 0.0f; mem1[i][j] = 0.0f; }
    }
    #pragma unroll
    for (int i = 0; i < 4; ++i) {
        #pragma unroll
        for (int j = 0; j < 8; ++j) mem2[i][j] = 0.0f;
    }

    #pragma unroll 1
    for (int t = 0; t < T_STEPS; ++t) {
        #pragma unroll
        for (int i = 0; i < (MTILE * D_IN) / THREADS; ++i) {
            int idx = tid + i * THREADS;
            int m = idx >> 8;
            int d = idx & (D_IN - 1);
            float xv = __builtin_nontemporal_load(
                &xin[((size_t)(b0 + m) * T_STEPS + t) * D_IN + d]);
            xs[m][d] = (_Float16)xv;
        }
        __syncthreads();

        #pragma unroll
        for (int np = 0; np < 4; ++np) {
            const int n0 = (wave * 8 + np * 2) * 16;
            v8f acc0, acc1;
            #pragma unroll
            for (int j = 0; j < 8; ++j) { acc0[j] = 0.0f; acc1[j] = 0.0f; }
            const _Float16* wr0 = W0h + (size_t)(n0 + lm) * D_IN + lhi * 16;
            const _Float16* wr1 = wr0 + (size_t)16 * D_IN;
            #pragma unroll 2
            for (int ks = 0; ks < D_IN; ks += 32) {
                v8h alo = *(const v8h*)&xs[lm][ks + lhi * 16];
                v8h ahi = *(const v8h*)&xs[lm][ks + lhi * 16 + 8];
                v16h a;
                #pragma unroll
                for (int j = 0; j < 8; ++j) { a[j] = alo[j]; a[j + 8] = ahi[j]; }
                v16h b0 = *(const v16h*)(wr0 + ks);
                v16h b1 = *(const v16h*)(wr1 + ks);
                acc0 = WMMA_F16(a, b0, acc0);
                acc1 = WMMA_F16(a, b1, acc1);
            }
            #pragma unroll
            for (int j = 0; j < 8; ++j) {
                float mv0 = mem0[np * 2][j] * 0.9f + acc0[j];
                float mv1 = mem0[np * 2 + 1][j] * 0.9f + acc1[j];
                bool sp0 = (mv0 >= 1.0f);
                bool sp1 = (mv1 >= 1.0f);
                mem0[np * 2][j]     = sp0 ? 0.0f : mv0;
                mem0[np * 2 + 1][j] = sp1 ? 0.0f : mv1;
                s0[j + 8 * lhi][n0 + lm]      = sp0 ? (unsigned char)1 : (unsigned char)0;
                s0[j + 8 * lhi][n0 + 16 + lm] = sp1 ? (unsigned char)1 : (unsigned char)0;
            }
        }
        __syncthreads();

        #pragma unroll
        for (int np = 0; np < 4; ++np) {
            const int n0 = (wave * 8 + np * 2) * 16;
            v8i acc0, acc1;
            #pragma unroll
            for (int j = 0; j < 8; ++j) { acc0[j] = 0; acc1[j] = 0; }
            const unsigned char* wr0 = W1b + (size_t)(n0 + lm) * H1 + lhi * 16;
            const unsigned char* wr1 = wr0 + (size_t)16 * H1;
            #pragma unroll 1
            for (int ks = 0; ks < H1; ks += 64) {
                const int koff = ks + lhi * 16;
                v8i a, b0, b1;
                LOAD_A8(a, s0[lm], koff);
                LOAD_B8(b0, wr0, ks);
                LOAD_B8(b1, wr1, ks);
                acc0 = WMMA_IU8(a, b0, acc0);
                acc1 = WMMA_IU8(a, b1, acc1);
            }
            #pragma unroll
            for (int j = 0; j < 8; ++j) {
                float mv0 = mem1[np * 2][j] * 0.9f + (float)acc0[j];
                float mv1 = mem1[np * 2 + 1][j] * 0.9f + (float)acc1[j];
                bool sp0 = (mv0 >= 1.0f);
                bool sp1 = (mv1 >= 1.0f);
                mem1[np * 2][j]     = sp0 ? 0.0f : mv0;
                mem1[np * 2 + 1][j] = sp1 ? 0.0f : mv1;
                s1[j + 8 * lhi][n0 + lm]      = sp0 ? (unsigned char)1 : (unsigned char)0;
                s1[j + 8 * lhi][n0 + 16 + lm] = sp1 ? (unsigned char)1 : (unsigned char)0;
            }
        }
        __syncthreads();

        #pragma unroll
        for (int np = 0; np < 2; ++np) {
            const int n0 = (wave * 4 + np * 2) * 16;
            v8i acc0, acc1;
            #pragma unroll
            for (int j = 0; j < 8; ++j) { acc0[j] = 0; acc1[j] = 0; }
            const unsigned char* wr0 = W2b + (size_t)(n0 + lm) * H2 + lhi * 16;
            const unsigned char* wr1 = wr0 + (size_t)16 * H2;
            #pragma unroll 1
            for (int ks = 0; ks < H2; ks += 64) {
                const int koff = ks + lhi * 16;
                v8i a, b0, b1;
                LOAD_A8(a, s1[lm], koff);
                LOAD_B8(b0, wr0, ks);
                LOAD_B8(b1, wr1, ks);
                acc0 = WMMA_IU8(a, b0, acc0);
                acc1 = WMMA_IU8(a, b1, acc1);
            }
            #pragma unroll
            for (int j = 0; j < 8; ++j) {
                float mv0 = mem2[np * 2][j] * 0.9f + (float)acc0[j];
                float mv1 = mem2[np * 2 + 1][j] * 0.9f + (float)acc1[j];
                bool sp0 = (mv0 >= 1.0f);
                bool sp1 = (mv1 >= 1.0f);
                mem2[np * 2][j]     = sp0 ? 0.0f : mv0;
                mem2[np * 2 + 1][j] = sp1 ? 0.0f : mv1;
                float* orow = &out[((size_t)(b0 + j + 8 * lhi) * T_STEPS + t) * D_OUT];
                __builtin_nontemporal_store(sp0 ? 1.0f : 0.0f, orow + n0 + lm);
                __builtin_nontemporal_store(sp1 ? 1.0f : 0.0f, orow + n0 + 16 + lm);
            }
        }
    }
}


extern "C" void kernel_launch(void* const* d_in, const int* in_sizes, int n_in,
                              void* d_out, int out_size, void* d_ws, size_t ws_size,
                              hipStream_t stream) {
    const float* x  = (const float*)d_in[0];
    const float* W0 = (const float*)d_in[1];
    const float* W1 = (const float*)d_in[2];
    const float* W2 = (const float*)d_in[3];
    float* out = (float*)d_out;

    const int B = in_sizes[0] / (T_STEPS * D_IN);

    char* ws = (char*)d_ws;
    _Float16*      W0h = (_Float16*)ws;
    unsigned char* W1b = (unsigned char*)(ws + (size_t)H1 * D_IN * sizeof(_Float16));
    unsigned char* W2b = W1b + (size_t)H2 * H1;

    {
        int n = H1 * D_IN;
        cvt_f32_to_f16_kernel<<<(n + 255) / 256, 256, 0, stream>>>(W0, W0h, n);
    }
    {
        int n = H2 * H1;
        cvt_f32_to_u8_kernel<<<(n + 255) / 256, 256, 0, stream>>>(W1, W1b, n);
    }
    {
        int n = D_OUT * H2;
        cvt_f32_to_u8_kernel<<<(n + 255) / 256, 256, 0, stream>>>(W2, W2b, n);
    }

    snn_fused_kernel<<<B / MTILE, THREADS, 0, stream>>>(x, W0h, W1b, W2b, out);
}
